// MixtureOfExperts_11836929868214
// MI455X (gfx1250) — matrix operand order repaired and checked by reading, not compiled, not run on hardware
//
#include <hip/hip_runtime.h>
#include <hip/hip_bf16.h>
#include <math.h>

typedef __bf16 bf16_t;
typedef __attribute__((ext_vector_type(16))) __bf16 v16bf;
typedef __attribute__((ext_vector_type(8)))  __bf16 v8bf;
typedef __attribute__((ext_vector_type(8)))  float  v8f;
typedef __attribute__((ext_vector_type(4)))  int    v4i;

#if defined(__AMDGCN__) && __has_builtin(__builtin_amdgcn_global_load_async_to_lds_b128)
#define HAS_ASYNC_LDS 1
typedef __attribute__((address_space(1))) v4i g_v4i; // global int4
typedef __attribute__((address_space(3))) v4i l_v4i; // LDS int4
#else
#define HAS_ASYNC_LDS 0
#endif

// ---------------- problem constants ----------------
constexpr int Bb = 4, Ll = 2048, Dd = 1024;
constexpr int Ee = 8, Kk = 2, Hh = 4096;
constexpr int T  = Bb * Ll;                  // 8192 tokens
// GEMM tiling
constexpr int BM = 128, BN = 128, BK = 32;
constexpr int SLOT_CAP   = T * Kk + Ee * BM; // 17408 (tile-aligned segments fit)
constexpr int SLOT_TILES = SLOT_CAP / BM;    // 136

// ---------------- workspace layout (bytes) ----------------
constexpr size_t OFF_XN   = 0;
constexpr size_t SZ_XN    = (size_t)T * Dd * 2;            // xn bf16
constexpr size_t OFF_W1T  = OFF_XN + SZ_XN;
constexpr size_t SZ_W1T   = (size_t)Ee * Hh * Dd * 2;      // W1^T bf16 [E][H][D]
constexpr size_t OFF_W2T  = OFF_W1T + SZ_W1T;
constexpr size_t SZ_W2T   = (size_t)Ee * Dd * Hh * 2;      // W2^T bf16 [E][D][H]
constexpr size_t OFF_H    = OFF_W2T + SZ_W2T;
constexpr size_t SZ_H     = (size_t)SLOT_CAP * Hh * 2;     // hidden bf16
constexpr size_t OFF_Y    = OFF_H + SZ_H;
constexpr size_t SZ_Y     = (size_t)SLOT_CAP * Dd * 4;     // y fp32
constexpr size_t OFF_TOPE = OFF_Y + SZ_Y;                  // int[2T]
constexpr size_t OFF_TOPG = OFF_TOPE + (size_t)2 * T * 4;  // float[2T]
constexpr size_t OFF_TSL  = OFF_TOPG + (size_t)2 * T * 4;  // int[2T] token->slot
constexpr size_t OFF_ROWT = OFF_TSL  + (size_t)2 * T * 4;  // int[SLOT_CAP] slot->token
constexpr size_t OFF_SEG  = OFF_ROWT + (size_t)SLOT_CAP * 4; // int[9]
constexpr size_t OFF_CNT  = OFF_SEG + 64;                  // int[8]
constexpr size_t OFF_CTR  = OFF_CNT + 64;                  // int[8]

// ---------------- helpers ----------------
__device__ __forceinline__ float block_reduce_add(float v, float* sbuf) {
  const int tid = threadIdx.x;
  sbuf[tid] = v; __syncthreads();
  #pragma unroll
  for (int s = 128; s > 0; s >>= 1) {
    if (tid < s) sbuf[tid] += sbuf[tid + s];
    __syncthreads();
  }
  float r = sbuf[0]; __syncthreads();
  return r;
}

__device__ __forceinline__ float gelu_exact(float v) {
  return 0.5f * v * (1.0f + erff(v * 0.70710678118654752440f));
}

__device__ __forceinline__ v16bf pack16(v8bf lo, v8bf hi) {
  v16bf r;
  #pragma unroll
  for (int i = 0; i < 8; ++i) { r[i] = lo[i]; r[i + 8] = hi[i]; }
  return r;
}

// 16-byte global -> LDS copy. Prefer CDNA5 async-to-LDS (ASYNCcnt, bypasses VGPRs).
__device__ __forceinline__ void copy16_g2l(const bf16_t* __restrict__ gsrc,
                                           bf16_t* __restrict__ ldst) {
#if HAS_ASYNC_LDS
  __builtin_amdgcn_global_load_async_to_lds_b128(
      (g_v4i*)gsrc, (l_v4i*)ldst, 0, 0);
#else
  *(v8bf*)ldst = *(const v8bf*)gsrc;
#endif
}

__device__ __forceinline__ void wait_g2l() {
#if HAS_ASYNC_LDS
#if __has_builtin(__builtin_amdgcn_s_wait_asynccnt)
  __builtin_amdgcn_s_wait_asynccnt(0);
#else
  asm volatile("s_wait_asynccnt 0x0" ::: "memory");
#endif
#endif
}

// ---------------- 0) transpose-convert fp32 [E][R][C] -> bf16 [E][C][R] ----------------
__global__ __launch_bounds__(256) void transpose_cvt_kernel(
    const float* __restrict__ in, bf16_t* __restrict__ out, int R, int C) {
  __shared__ float tile[32][33];
  const int e  = blockIdx.z;
  const int c0 = blockIdx.x * 32, r0 = blockIdx.y * 32;
  const int tx = threadIdx.x, ty = threadIdx.y; // (32, 8)
  const float* pin  = in  + (size_t)e * R * C;
  bf16_t*      pout = out + (size_t)e * C * R;
  #pragma unroll
  for (int i = 0; i < 32; i += 8)
    tile[ty + i][tx] = pin[(size_t)(r0 + ty + i) * C + (c0 + tx)];
  __syncthreads();
  #pragma unroll
  for (int i = 0; i < 32; i += 8)
    pout[(size_t)(c0 + ty + i) * R + (r0 + tx)] = (bf16_t)tile[tx][ty + i];
}

// ---------------- 1) init slot->token map ----------------
__global__ void init_rows_kernel(int* __restrict__ row_token) {
  int i = blockIdx.x * 256 + threadIdx.x;
  if (i < SLOT_CAP) row_token[i] = 0;
}

// ---------------- 2) fused LayerNorm + gating (one block per token) ----------------
__global__ __launch_bounds__(256) void ln_gate_kernel(
    const float* __restrict__ x, const float* __restrict__ gamma,
    const float* __restrict__ beta, const float* __restrict__ gate_w,
    bf16_t* __restrict__ xnbf, int* __restrict__ top_e, float* __restrict__ top_g) {
  __shared__ float sbuf[256];
  __shared__ float lg[Ee];
  const int t = blockIdx.x, tid = threadIdx.x;
  const float4 xv = *(const float4*)(x + (size_t)t * Dd + tid * 4);
  float v[4] = {xv.x, xv.y, xv.z, xv.w};
  float s = 0.f, ss = 0.f;
  #pragma unroll
  for (int i = 0; i < 4; ++i) { s += v[i]; ss += v[i] * v[i]; }
  const float tot  = block_reduce_add(s,  sbuf);
  const float tot2 = block_reduce_add(ss, sbuf);
  const float mu  = tot * (1.0f / Dd);
  const float var = tot2 * (1.0f / Dd) - mu * mu;   // biased variance
  const float rn  = rsqrtf(var + 1e-5f);
  float la[Ee];
  #pragma unroll
  for (int e = 0; e < Ee; ++e) la[e] = 0.f;
  #pragma unroll
  for (int i = 0; i < 4; ++i) {
    const int d = tid * 4 + i;
    const float xn = (v[i] - mu) * rn * gamma[d] + beta[d];
    xnbf[(size_t)t * Dd + d] = (bf16_t)xn;
    #pragma unroll
    for (int e = 0; e < Ee; ++e) la[e] += xn * gate_w[e * Dd + d];
  }
  for (int e = 0; e < Ee; ++e) {
    const float r = block_reduce_add(la[e], sbuf);
    if (tid == 0) lg[e] = r;
  }
  __syncthreads();
  if (tid == 0) {
    int e0 = 0; float v0 = lg[0];
    #pragma unroll
    for (int e = 1; e < Ee; ++e) if (lg[e] > v0) { v0 = lg[e]; e0 = e; }
    int e1 = -1; float v1 = -3.4e38f;
    #pragma unroll
    for (int e = 0; e < Ee; ++e) if (e != e0 && lg[e] > v1) { v1 = lg[e]; e1 = e; }
    const float ga = 1.0f / (1.0f + expf(v1 - v0)); // top-2 softmax
    top_e[2 * t] = e0; top_e[2 * t + 1] = e1;
    top_g[2 * t] = ga; top_g[2 * t + 1] = 1.0f - ga;
  }
}

// ---------------- 3) counts, tile-aligned segment offsets, bal_loss ----------------
__global__ __launch_bounds__(256) void finalize_kernel(
    const int* __restrict__ top_e, const float* __restrict__ top_g,
    int* __restrict__ seg_start, int* __restrict__ counts, int* __restrict__ ctr,
    float* __restrict__ loss_out) {
  const int tid = threadIdx.x;
  float ls[Ee]; int lc[Ee];
  #pragma unroll
  for (int e = 0; e < Ee; ++e) { ls[e] = 0.f; lc[e] = 0; }
  for (int i = tid; i < 2 * T; i += 256) {
    const int e = top_e[i]; const float g = top_g[i];
    #pragma unroll
    for (int e2 = 0; e2 < Ee; ++e2) {
      const bool m = (e == e2);
      ls[e2] += m ? g : 0.f;
      lc[e2] += m ? 1 : 0;
    }
  }
  __shared__ float fbuf[256]; __shared__ int ibuf[256];
  __shared__ float lsum[Ee]; __shared__ int lcnt[Ee];
  for (int e = 0; e < Ee; ++e) {
    fbuf[tid] = ls[e]; ibuf[tid] = lc[e]; __syncthreads();
    #pragma unroll
    for (int s2 = 128; s2 > 0; s2 >>= 1) {
      if (tid < s2) { fbuf[tid] += fbuf[tid + s2]; ibuf[tid] += ibuf[tid + s2]; }
      __syncthreads();
    }
    if (tid == 0) { lsum[e] = fbuf[0]; lcnt[e] = ibuf[0]; }
    __syncthreads();
  }
  if (tid == 0) {
    int s = 0; float loss = 0.f;
    for (int e = 0; e < Ee; ++e) {
      counts[e] = lcnt[e]; ctr[e] = 0; seg_start[e] = s;
      s += ((lcnt[e] + BM - 1) / BM) * BM;
      const float d = lsum[e] / (float)T - 1.0f / (float)Ee;
      loss += d * d;
    }
    seg_start[Ee] = s;
    loss_out[0] = loss / (float)Ee;
  }
}

// ---------------- 4) scatter token -> expert slots ----------------
__global__ void scatter_kernel(const int* __restrict__ top_e,
                               const int* __restrict__ seg_start, int* __restrict__ ctr,
                               int* __restrict__ row_token, int* __restrict__ token_slot) {
  const int t = blockIdx.x * 256 + threadIdx.x;
  if (t >= T) return;
  #pragma unroll
  for (int k = 0; k < 2; ++k) {
    const int e = top_e[2 * t + k];
    const int pos = atomicAdd(&ctr[e], 1);
    const int slot = seg_start[e] + pos;
    row_token[slot] = t;
    token_slot[2 * t + k] = slot;
  }
}

// ---------------- 5/6) grouped WMMA GEMM (bf16 -> f32 acc) ----------------
// BM=128 x BN=128 x BK=32, 256 threads = 8 waves, wave grid 2(m) x 4(n),
// each wave owns a 64x32 patch = 4x2 WMMA 16x16 tiles.
// Double-buffered LDS staged via async-to-LDS (ASYNCcnt) when available.
template <bool GATHER, bool GELU_EPI>
__global__ __launch_bounds__(256) void moe_gemm_kernel(
    const bf16_t* __restrict__ Abase,   // GATHER: xn [T][Kdim]; else h [SLOT_CAP][Kdim]
    const bf16_t* __restrict__ Bbase,   // [E][Ndim][Kdim] (N-major, pre-transposed)
    const float*  __restrict__ bias,    // [E][Ndim] or nullptr
    const int*    __restrict__ row_token,
    const int*    __restrict__ seg_start, const int* __restrict__ counts,
    void* __restrict__ OutP, int Kdim, int Ndim) {
  const int tileM0 = blockIdx.x * BM;
  int e = -1;
  #pragma unroll
  for (int i = 0; i < Ee; ++i)
    if (tileM0 >= seg_start[i] && tileM0 < seg_start[i + 1]) e = i;
  if (e < 0) return;                               // past last segment
  if (tileM0 >= seg_start[e] + counts[e]) return;  // all-padding tile

  const int n0 = blockIdx.y * BN;
  __shared__ __align__(16) bf16_t As[2][BM][BK + 8];
  __shared__ __align__(16) bf16_t Bs[2][BN][BK + 8];
  const bf16_t* Bp = Bbase + (size_t)e * Ndim * Kdim;

  const int tid  = threadIdx.x;
  const int lane = tid & 31, wave = tid >> 5;
  const int wm = wave >> 2;        // 0..1 -> 64 rows
  const int wn = wave & 3;         // 0..3 -> 32 cols
  const int lr  = tid >> 2;        // 0..63 staging row
  const int lcv = (tid & 3) * 8;   // staging col: 0,8,16,24

  size_t arow0, arow1;
  if (GATHER) {
    arow0 = (size_t)row_token[tileM0 + lr]      * (size_t)Kdim;
    arow1 = (size_t)row_token[tileM0 + lr + 64] * (size_t)Kdim;
  } else {
    arow0 = (size_t)(tileM0 + lr)      * (size_t)Kdim;
    arow1 = (size_t)(tileM0 + lr + 64) * (size_t)Kdim;
  }
  const size_t brow0 = (size_t)(n0 + lr)      * (size_t)Kdim;
  const size_t brow1 = (size_t)(n0 + lr + 64) * (size_t)Kdim;

  v8f acc[4][2];
  #pragma unroll
  for (int tm = 0; tm < 4; ++tm)
    #pragma unroll
    for (int tn = 0; tn < 2; ++tn)
      #pragma unroll
      for (int i = 0; i < 8; ++i) acc[tm][tn][i] = 0.f;

  // A frag: lanes 0-15 -> M=lane, K {0..7,16..23}; lanes 16-31 -> M=lane-16, K {8..15,24..31}
  // B frag (N-major LDS): lane n -> col n&15, K {0..7,16..23} / {8..15,24..31}
  const int mrow = lane & 15;
  const int kb   = (lane >> 4) * 8;
  const int kb2  = (lane >> 4) * 16;

  const int niter = Kdim / BK;
  // prologue: stage buffer 0
  {
    const int k0 = 0;
    copy16_g2l(&Abase[arow0 + k0 + lcv], &As[0][lr][lcv]);
    copy16_g2l(&Abase[arow1 + k0 + lcv], &As[0][lr + 64][lcv]);
    copy16_g2l(&Bp[brow0 + k0 + lcv],    &Bs[0][lr][lcv]);
    copy16_g2l(&Bp[brow1 + k0 + lcv],    &Bs[0][lr + 64][lcv]);
  }
  for (int i = 0; i < niter; ++i) {
    const int cur = i & 1;
    wait_g2l();          // own async copies into `cur` landed in LDS
    __syncthreads();     // everyone's copies visible; everyone done reading cur^1
    if (i + 1 < niter) { // stage next tile into the other buffer
      const int k0 = (i + 1) * BK;
      const int nxt = cur ^ 1;
      copy16_g2l(&Abase[arow0 + k0 + lcv], &As[nxt][lr][lcv]);
      copy16_g2l(&Abase[arow1 + k0 + lcv], &As[nxt][lr + 64][lcv]);
      copy16_g2l(&Bp[brow0 + k0 + lcv],    &Bs[nxt][lr][lcv]);
      copy16_g2l(&Bp[brow1 + k0 + lcv],    &Bs[nxt][lr + 64][lcv]);
    }
    v16bf a[4], b[2];
    #pragma unroll
    for (int tm = 0; tm < 4; ++tm) {
      const int row = wm * 64 + tm * 16 + mrow;
      a[tm] = pack16(*(const v8bf*)&As[cur][row][kb],
                     *(const v8bf*)&As[cur][row][kb + 16]);
    }
    #pragma unroll
    for (int tn = 0; tn < 2; ++tn) {
      const int col = wn * 32 + tn * 16 + mrow;
      b[tn] = pack16(*(const v8bf*)&Bs[cur][col][kb],
                     *(const v8bf*)&Bs[cur][col][kb + 16]);
    }
    #pragma unroll
    for (int tm = 0; tm < 4; ++tm)
      #pragma unroll
      for (int tn = 0; tn < 2; ++tn)
        acc[tm][tn] = __builtin_amdgcn_wmma_f32_16x16x32_bf16(
            false, a[tm], false, b[tn], (short)0, acc[tm][tn], false, false);
  }

  // epilogue: C/D layout -> N = lane&15, M = vgpr + 8*(lane>=16)
  const int nloc = lane & 15;
  const int mhi  = (lane >> 4) * 8;
  #pragma unroll
  for (int tm = 0; tm < 4; ++tm) {
    #pragma unroll
    for (int tn = 0; tn < 2; ++tn) {
      const int ncol = n0 + wn * 32 + tn * 16 + nloc;
      #pragma unroll
      for (int v2 = 0; v2 < 8; ++v2) {
        const int slot = tileM0 + wm * 64 + tm * 16 + mhi + v2;
        float val = acc[tm][tn][v2];
        if (GELU_EPI) {
          val += bias[(size_t)e * Ndim + ncol];
          val = gelu_exact(val);
          ((bf16_t*)OutP)[(size_t)slot * Ndim + ncol] = (bf16_t)val;
        } else {
          ((float*)OutP)[(size_t)slot * Ndim + ncol] = val;
        }
      }
    }
  }
}

// ---------------- 7) combine: out = x + g0*(y0+b2[e0]) + g1*(y1+b2[e1]) ----------------
__global__ __launch_bounds__(256) void combine_kernel(
    const float* __restrict__ x, const float* __restrict__ b2, const float* __restrict__ y,
    const int* __restrict__ token_slot, const int* __restrict__ top_e,
    const float* __restrict__ top_g, float* __restrict__ out) {
  const int t = blockIdx.x, tid = threadIdx.x;
  const int s0 = token_slot[2 * t], s1 = token_slot[2 * t + 1];
  const int e0 = top_e[2 * t], e1 = top_e[2 * t + 1];
  const float g0 = top_g[2 * t], g1 = top_g[2 * t + 1];
  const size_t base = (size_t)t * Dd + tid * 4;
  const float4 xv  = *(const float4*)(x + base);
  const float4 y0  = *(const float4*)(y + (size_t)s0 * Dd + tid * 4);
  const float4 y1  = *(const float4*)(y + (size_t)s1 * Dd + tid * 4);
  const float4 bb0 = *(const float4*)(b2 + (size_t)e0 * Dd + tid * 4);
  const float4 bb1 = *(const float4*)(b2 + (size_t)e1 * Dd + tid * 4);
  float4 o;
  o.x = xv.x + g0 * (y0.x + bb0.x) + g1 * (y1.x + bb1.x);
  o.y = xv.y + g0 * (y0.y + bb0.y) + g1 * (y1.y + bb1.y);
  o.z = xv.z + g0 * (y0.z + bb0.z) + g1 * (y1.z + bb1.z);
  o.w = xv.w + g0 * (y0.w + bb0.w) + g1 * (y1.w + bb1.w);
  *(float4*)(out + base) = o;
}

// ---------------- launcher ----------------
extern "C" void kernel_launch(void* const* d_in, const int* in_sizes, int n_in,
                              void* d_out, int out_size, void* d_ws, size_t ws_size,
                              hipStream_t stream) {
  const float* x      = (const float*)d_in[0];
  const float* gamma  = (const float*)d_in[1];
  const float* beta   = (const float*)d_in[2];
  const float* gate_w = (const float*)d_in[3];
  const float* W1     = (const float*)d_in[4];
  const float* b1     = (const float*)d_in[5];
  const float* W2     = (const float*)d_in[6];
  const float* b2     = (const float*)d_in[7];
  float* out = (float*)d_out;

  char* ws = (char*)d_ws;
  bf16_t* xnbf      = (bf16_t*)(ws + OFF_XN);
  bf16_t* W1T       = (bf16_t*)(ws + OFF_W1T);
  bf16_t* W2T       = (bf16_t*)(ws + OFF_W2T);
  bf16_t* h_buf     = (bf16_t*)(ws + OFF_H);
  float*  y_buf     = (float*) (ws + OFF_Y);
  int*    top_e     = (int*)   (ws + OFF_TOPE);
  float*  top_g     = (float*) (ws + OFF_TOPG);
  int*    token_slot= (int*)   (ws + OFF_TSL);
  int*    row_token = (int*)   (ws + OFF_ROWT);
  int*    seg_start = (int*)   (ws + OFF_SEG);
  int*    counts    = (int*)   (ws + OFF_CNT);
  int*    ctr       = (int*)   (ws + OFF_CTR);

  // weight fp32 -> bf16 with transpose to N-major (coalesced both sides)
  transpose_cvt_kernel<<<dim3(Hh / 32, Dd / 32, Ee), dim3(32, 8), 0, stream>>>(W1, W1T, Dd, Hh);
  transpose_cvt_kernel<<<dim3(Dd / 32, Hh / 32, Ee), dim3(32, 8), 0, stream>>>(W2, W2T, Hh, Dd);

  init_rows_kernel<<<(SLOT_CAP + 255) / 256, 256, 0, stream>>>(row_token);
  ln_gate_kernel<<<T, 256, 0, stream>>>(x, gamma, beta, gate_w, xnbf, top_e, top_g);
  finalize_kernel<<<1, 256, 0, stream>>>(top_e, top_g, seg_start, counts, ctr,
                                         out + (size_t)T * Dd); // bal_loss slot
  scatter_kernel<<<(T + 255) / 256, 256, 0, stream>>>(top_e, seg_start, ctr,
                                                      row_token, token_slot);
  // GEMM1: h = gelu(X_e @ W1[e] + b1[e]);   A gathered, K=1024, N=4096
  moe_gemm_kernel<true, true><<<dim3(SLOT_TILES, Hh / BN), 256, 0, stream>>>(
      xnbf, W1T, b1, row_token, seg_start, counts, (void*)h_buf, Dd, Hh);
  // GEMM2: y = H_e @ W2[e];                 A contiguous slots, K=4096, N=1024
  moe_gemm_kernel<false, false><<<dim3(SLOT_TILES, Dd / BN), 256, 0, stream>>>(
      h_buf, W2T, nullptr, row_token, seg_start, counts, (void*)y_buf, Hh, Dd);
  combine_kernel<<<T, 256, 0, stream>>>(x, b2, y_buf, token_slot, top_e, top_g, out);
}
